// GATNet_6751688589607
// MI455X (gfx1250) — matrix operand order repaired and checked by reading, not compiled, not run on hardware
//
#include <hip/hip_runtime.h>
#include <cstddef>
#include <cstdint>

// ---------------- problem constants (match reference) ----------------
#define NN_NODES 65536
#define N_EDGES  524288
#define F_IN     128
#define HEADS    4
#define C1T      32          // layer-1 per-head channels
#define C2T      8           // layer-2 per-head channels
#define D1       (HEADS*C1T) // 128
#define D2       (HEADS*C2T) // 32
#define NEG_SLOPE 0.2f

typedef __attribute__((ext_vector_type(16))) __bf16 bf16x16;
typedef __attribute__((ext_vector_type(8)))  __bf16 bf16x8;
typedef __attribute__((ext_vector_type(8)))  float  f32x8;

// ---------------- helpers ----------------
__device__ __forceinline__ __bf16 f2bf(float f) {
    union { float f; unsigned u; } in; in.f = f;
    unsigned u = in.u;
    unsigned r = (u + 0x7fffu + ((u >> 16) & 1u)) >> 16;   // round-to-nearest-even
    union { unsigned short s; __bf16 b; } out; out.s = (unsigned short)r;
    return out.b;
}
// order-preserving float<->uint for atomicMax on floats
__device__ __forceinline__ unsigned enc_ord(float f) {
    unsigned u = __float_as_uint(f);
    return (u & 0x80000000u) ? ~u : (u | 0x80000000u);
}
__device__ __forceinline__ float dec_ord(unsigned e) {
    unsigned u = (e & 0x80000000u) ? (e & 0x7fffffffu) : ~e;
    return __uint_as_float(u);
}

// ---------------- fills / converts ----------------
__global__ void fill_u32(unsigned* __restrict__ p, unsigned v, size_t n) {
    size_t i = (size_t)blockIdx.x * blockDim.x + threadIdx.x;
    size_t stride = (size_t)gridDim.x * blockDim.x;
    for (; i < n; i += stride) p[i] = v;
}

__global__ void cvt_f32_bf16(const float* __restrict__ src, __bf16* __restrict__ dst, size_t n) {
    size_t i = (size_t)blockIdx.x * blockDim.x + threadIdx.x;
    size_t stride = (size_t)gridDim.x * blockDim.x;
    for (; i < n; i += stride) dst[i] = f2bf(src[i]);
}

// W [K,N] f32 row-major  ->  Wt [N,K] bf16 row-major (so B fragments are contiguous in K)
__global__ void cvt_w_transpose(const float* __restrict__ W, __bf16* __restrict__ Wt, int K, int N) {
    int idx = blockIdx.x * blockDim.x + threadIdx.x;
    if (idx >= K * N) return;
    int k = idx / N, n = idx - k * N;
    Wt[(size_t)n * K + k] = f2bf(W[(size_t)k * N + n]);
}

// ---------------- WMMA GEMM: Out[M,N] = A[M,K](bf16) * Bt[N,K]^T(bf16) + bias ----------------
// one wave computes one 16x16 f32 tile; K stepped by 32 via v_wmma_f32_16x16x32_bf16.
__global__ void gemm_bf16_wmma(const __bf16* __restrict__ A,   // [M,K] bf16
                               const __bf16* __restrict__ Bt,  // [N,K] bf16 (B transposed)
                               const float*  __restrict__ bias,// [N]
                               float* __restrict__ Out,        // [M,N] f32
                               int M, int N, int K)
{
    int waveId = (int)((blockIdx.x * blockDim.x + threadIdx.x) >> 5);
    int lane   = threadIdx.x & 31;
    int tilesN = N >> 4;
    int tilesM = M >> 4;
    if (waveId >= tilesM * tilesN) return;
    int tm = waveId / tilesN;
    int tn = waveId - tm * tilesN;

    int hi = lane >> 4;     // lane half: 0 or 1
    int r  = lane & 15;

    // A fragment base: row M = tm*16 + r ; lane-half selects K sub-offset (+0 / +8)
    const __bf16* aBase = A  + (size_t)(tm * 16 + r) * K + hi * 8;
    // B fragment base: col N = tn*16 + r ; lane-half selects K sub-offset (+0 / +8)
    const __bf16* bBase = Bt + (size_t)(tn * 16 + r) * K + hi * 8;

    f32x8 acc = {};
    for (int k0 = 0; k0 < K; k0 += 32) {
        bf16x8 a_lo = *(const bf16x8*)(aBase + k0);        // K = k0+hi*8 + 0..7
        bf16x8 a_hi = *(const bf16x8*)(aBase + k0 + 16);   // K = k0+hi*8 + 16..23
        bf16x8 b_lo = *(const bf16x8*)(bBase + k0);        // K = k0+hi*8 + 0..7
        bf16x8 b_hi = *(const bf16x8*)(bBase + k0 + 16);   // K = k0+hi*8 + 16..23
        bf16x16 a, b;
#pragma unroll
        for (int i = 0; i < 8; ++i) {
            a[i] = a_lo[i]; a[8 + i] = a_hi[i];
            b[i] = b_lo[i]; b[8 + i] = b_hi[i];
        }
        acc = __builtin_amdgcn_wmma_f32_16x16x32_bf16(
                  false, a, false, b, (short)0, acc, false, false);
    }

    int n = tn * 16 + r;
    float bv = bias[n];
#pragma unroll
    for (int i = 0; i < 8; ++i) {
        int m = tm * 16 + hi * 8 + i;    // VGPR i -> M = i (lanes 0-15) / 8+i (lanes 16-31)
        Out[(size_t)m * N + n] = acc[i] + bv;
    }
}

// ---------------- edge phase ----------------
// logits[e,h] = sum_c att[h,c] * leaky_relu(xl[src,h,c] + xr[dst,h,c]); atomic segment-max into mmax
template<int H, int C>
__global__ void edge_logits_k(const float* __restrict__ xl, const float* __restrict__ xr,
                              const long long* __restrict__ src, const long long* __restrict__ dst,
                              const float* __restrict__ att,
                              float* __restrict__ logits, unsigned* __restrict__ mmax, int E)
{
    int idx = blockIdx.x * blockDim.x + threadIdx.x;   // one thread per (edge, head)
    if (idx >= E * H) return;
    int e = idx / H, h = idx - e * H;
    size_t s = (size_t)src[e];
    size_t d = (size_t)dst[e];
    const float* pl = xl + s * (H * C) + h * C;
    const float* pr = xr + d * (H * C) + h * C;
    const float* pa = att + h * C;
    float acc = 0.f;
#pragma unroll
    for (int c = 0; c < C; ++c) {
        float v = pl[c] + pr[c];
        v = v > 0.f ? v : NEG_SLOPE * v;
        acc += pa[c] * v;
    }
    logits[idx] = acc;
    atomicMax(&mmax[d * H + h], enc_ord(acc));
}

// p = exp(logit - m[dst]); logits <- p; denom[dst] += p
template<int H>
__global__ void edge_expdenom_k(const long long* __restrict__ dst,
                                float* __restrict__ logits,
                                const unsigned* __restrict__ mmax,
                                float* __restrict__ denom, int E)
{
    int idx = blockIdx.x * blockDim.x + threadIdx.x;
    if (idx >= E * H) return;
    int e = idx / H, h = idx - e * H;
    size_t d = (size_t)dst[e];
    float m = dec_ord(mmax[d * H + h]);
    float p = expf(logits[idx] - m);
    logits[idx] = p;
    atomicAdd(&denom[d * H + h], p);
}

// out[dst,h,:] += xl[src,h,:] * (p / (denom[dst,h] + eps))
template<int H, int C>
__global__ void edge_scatter_k(const long long* __restrict__ src, const long long* __restrict__ dst,
                               const float* __restrict__ xl, const float* __restrict__ p,
                               const float* __restrict__ denom,
                               float* __restrict__ out, int E)
{
    int idx = blockIdx.x * blockDim.x + threadIdx.x;
    if (idx >= E * H) return;
    int e = idx / H, h = idx - e * H;
    size_t s = (size_t)src[e];
    size_t d = (size_t)dst[e];
    float alpha = p[idx] / (denom[d * H + h] + 1e-16f);
    const float* pl = xl  + s * (H * C) + h * C;
    float*       po = out + d * (H * C) + h * C;
#pragma unroll
    for (int c = 0; c < C; ++c) atomicAdd(&po[c], pl[c] * alpha);
}

// v = elu(acc + bias); write f32 (possibly in-place or d_out) and optional bf16 copy for next GEMM
__global__ void epilogue_k(const float* __restrict__ acc, const float* __restrict__ bias,
                           float* __restrict__ hf, __bf16* __restrict__ hb, int nNodes, int D)
{
    int idx = blockIdx.x * blockDim.x + threadIdx.x;
    if (idx >= nNodes * D) return;
    int f = idx % D;
    float v = acc[idx] + bias[f];
    v = v > 0.f ? v : (expf(v) - 1.f);
    hf[idx] = v;
    if (hb) hb[idx] = f2bf(v);
}

// ---------------- launch ----------------
static inline int gb(long long n, int b) { return (int)((n + b - 1) / b); }

extern "C" void kernel_launch(void* const* d_in, const int* in_sizes, int n_in,
                              void* d_out, int out_size, void* d_ws, size_t ws_size,
                              hipStream_t stream)
{
    const float*     x     = (const float*)d_in[0];
    const long long* ei    = (const long long*)d_in[1];
    const float*     Wl1   = (const float*)d_in[2];
    const float*     bl1   = (const float*)d_in[3];
    const float*     Wr1   = (const float*)d_in[4];
    const float*     br1   = (const float*)d_in[5];
    const float*     att1  = (const float*)d_in[6];
    const float*     bias1 = (const float*)d_in[7];
    const float*     Wl2   = (const float*)d_in[8];
    const float*     bl2   = (const float*)d_in[9];
    const float*     Wr2   = (const float*)d_in[10];
    const float*     br2   = (const float*)d_in[11];
    const float*     att2  = (const float*)d_in[12];
    const float*     bias2 = (const float*)d_in[13];
    float* out = (float*)d_out;

    const long long* srcIdx = ei;            // edge_index[0]
    const long long* dstIdx = ei + N_EDGES;  // edge_index[1]

    // ---- workspace layout (256B aligned slabs) ----
    char* w = (char*)d_ws;
    size_t off = 0;
    auto take = [&](size_t bytes) -> void* {
        void* p = w + off;
        off = (off + bytes + 255) & ~(size_t)255;
        return p;
    };
    __bf16* xb    = (__bf16*)take((size_t)NN_NODES * F_IN * 2);  // x bf16, reused as h1 bf16
    __bf16* Wl1t  = (__bf16*)take((size_t)F_IN * D1 * 2);
    __bf16* Wr1t  = (__bf16*)take((size_t)F_IN * D1 * 2);
    __bf16* Wl2t  = (__bf16*)take((size_t)D1 * D2 * 2);
    __bf16* Wr2t  = (__bf16*)take((size_t)D1 * D2 * 2);
    float*  xl1   = (float*) take((size_t)NN_NODES * D1 * 4);    // reused as xl2
    float*  xr1   = (float*) take((size_t)NN_NODES * D1 * 4);    // reused as xr2
    float*  acc1  = (float*) take((size_t)NN_NODES * D1 * 4);    // reused as acc2
    float*  logit = (float*) take((size_t)N_EDGES * HEADS * 4);
    unsigned* mmax= (unsigned*)take((size_t)NN_NODES * HEADS * 4);
    float*  denom = (float*) take((size_t)NN_NODES * HEADS * 4);
    float*  xl2   = xl1;
    float*  xr2   = xr1;
    float*  acc2  = acc1;
    (void)ws_size; (void)in_sizes; (void)n_in; (void)out_size;

    const int B = 256;
    const int EH = N_EDGES * HEADS;

    // ---- convert inputs to bf16 (weights transposed) ----
    cvt_f32_bf16<<<gb((long long)NN_NODES * F_IN, B), B, 0, stream>>>(x, xb, (size_t)NN_NODES * F_IN);
    cvt_w_transpose<<<gb(F_IN * D1, B), B, 0, stream>>>(Wl1, Wl1t, F_IN, D1);
    cvt_w_transpose<<<gb(F_IN * D1, B), B, 0, stream>>>(Wr1, Wr1t, F_IN, D1);
    cvt_w_transpose<<<gb(D1 * D2, B), B, 0, stream>>>(Wl2, Wl2t, D1, D2);
    cvt_w_transpose<<<gb(D1 * D2, B), B, 0, stream>>>(Wr2, Wr2t, D1, D2);

    // ================= Layer 1 =================
    {
        long long waves = (long long)(NN_NODES / 16) * (D1 / 16);
        int blocks = gb(waves * 32, B);
        gemm_bf16_wmma<<<blocks, B, 0, stream>>>(xb, Wl1t, bl1, xl1, NN_NODES, D1, F_IN);
        gemm_bf16_wmma<<<blocks, B, 0, stream>>>(xb, Wr1t, br1, xr1, NN_NODES, D1, F_IN);
    }
    fill_u32<<<1024, B, 0, stream>>>(mmax, 0u, (size_t)NN_NODES * HEADS);
    fill_u32<<<1024, B, 0, stream>>>((unsigned*)denom, 0u, (size_t)NN_NODES * HEADS);
    fill_u32<<<4096, B, 0, stream>>>((unsigned*)acc1, 0u, (size_t)NN_NODES * D1);

    edge_logits_k<HEADS, C1T><<<gb(EH, B), B, 0, stream>>>(xl1, xr1, srcIdx, dstIdx, att1, logit, mmax, N_EDGES);
    edge_expdenom_k<HEADS><<<gb(EH, B), B, 0, stream>>>(dstIdx, logit, mmax, denom, N_EDGES);
    edge_scatter_k<HEADS, C1T><<<gb(EH, B), B, 0, stream>>>(srcIdx, dstIdx, xl1, logit, denom, acc1, N_EDGES);
    // elu(acc1 + bias1) -> acc1 (f32, in place) and xb (bf16 for layer-2 GEMM)
    epilogue_k<<<gb((long long)NN_NODES * D1, B), B, 0, stream>>>(acc1, bias1, acc1, xb, NN_NODES, D1);

    // ================= Layer 2 =================
    {
        long long waves = (long long)(NN_NODES / 16) * (D2 / 16);
        int blocks = gb(waves * 32, B);
        gemm_bf16_wmma<<<blocks, B, 0, stream>>>(xb, Wl2t, bl2, xl2, NN_NODES, D2, D1);
        gemm_bf16_wmma<<<blocks, B, 0, stream>>>(xb, Wr2t, br2, xr2, NN_NODES, D2, D1);
    }
    fill_u32<<<1024, B, 0, stream>>>(mmax, 0u, (size_t)NN_NODES * HEADS);
    fill_u32<<<1024, B, 0, stream>>>((unsigned*)denom, 0u, (size_t)NN_NODES * HEADS);
    fill_u32<<<1024, B, 0, stream>>>((unsigned*)acc2, 0u, (size_t)NN_NODES * D2);

    edge_logits_k<HEADS, C2T><<<gb(EH, B), B, 0, stream>>>(xl2, xr2, srcIdx, dstIdx, att2, logit, mmax, N_EDGES);
    edge_expdenom_k<HEADS><<<gb(EH, B), B, 0, stream>>>(dstIdx, logit, mmax, denom, N_EDGES);
    edge_scatter_k<HEADS, C2T><<<gb(EH, B), B, 0, stream>>>(srcIdx, dstIdx, xl2, logit, denom, acc2, N_EDGES);
    // final: elu(acc2 + bias2) -> d_out (f32), no bf16 copy
    epilogue_k<<<gb((long long)NN_NODES * D2, B), B, 0, stream>>>(acc2, bias2, out, (/*hb=*/(__bf16*)nullptr), NN_NODES, D2);
}
